// TitansBlock_30477087932976
// MI455X (gfx1250) — compile-verified, matrix operand order checked by reading, not run on hardware
//
#include <hip/hip_runtime.h>
#include <hip/hip_bf16.h>

// ---------------------------------------------------------------------------
// Titans block for MI455X (gfx1250).  Compute-bound => all matmuls on
// v_wmma_f32_16x16x32_bf16 with fp32 accumulation.  All GEMM operands are
// staged as bf16 so each 16x32 fragment is two global_load_b128 with zero
// converts in the inner loop; wave tile is 32x64 (8 WMMAs per K-chunk).
// Attention is flash-style (online softmax, P re-layout through LDS ordered
// by s_wait_dscnt, WMMA for QK^T and PV).
// ---------------------------------------------------------------------------

#define B_      8
#define T_      1024
#define C_      768
#define H_      12
#define DH_     64
#define HID_    1536
#define MM_     16
#define PP_     4
#define PREFIX_ 20
#define TA_     1044       // T_ + PREFIX_
#define LR_     0.01f
#define MU_     0.9f
#define DECAY_  0.001f

typedef __bf16 bf16_t;
typedef __attribute__((ext_vector_type(16))) __bf16 v16bf;
typedef __attribute__((ext_vector_type(8)))  __bf16 v8bf;
typedef __attribute__((ext_vector_type(8)))  float  v8f;

enum { EPI_NONE = 0, EPI_SILU = 1, EPI_GELU = 2, EPI_RES = 3, EPI_DPRED = 4, EPI_DSILU = 5 };

__device__ __forceinline__ v8f wmma_bf16(v16bf a, v16bf b, v8f c) {
  return __builtin_amdgcn_wmma_f32_16x16x32_bf16(false, a, false, b, (short)0, c, false, false);
}

__device__ __forceinline__ float siluf(float x) {
  float s = 1.0f / (1.0f + __expf(-x));
  return x * s;
}
__device__ __forceinline__ float dsiluf(float x) {
  float s = 1.0f / (1.0f + __expf(-x));
  return s * (1.0f + x * (1.0f - s));
}
__device__ __forceinline__ float geluf(float x) {
  return 0.5f * x * (1.0f + erff(x * 0.70710678118654752f));
}

// 16x32 bf16 A/B fragment (layout per CDNA5 ISA).
// trans=false: elem(r,k) = p[r*ld + k]  -> two 16-byte vector loads per lane.
// trans=true : elem(r,k) = p[k*ld + r]  -> strided scalar bf16 loads (k
//              clamped to klimit for the attention V fragment).
__device__ __forceinline__ v16bf load_frag_bf(const bf16_t* __restrict__ p, long ld,
                                              int rbase, int kbase,
                                              int rlimit, int klimit, bool trans) {
  int lane = threadIdx.x & 31;
  int rl = lane & 15, hi = lane >> 4;
  int r = rbase + rl;
  if (r > rlimit - 1) r = rlimit - 1;
  if (!trans) {
    const bf16_t* rp = p + (long)r * ld + kbase + (hi << 3);
    v8bf lo = *(const v8bf*)rp;         // k = kbase+hi*8 .. +7   (global_load_b128)
    v8bf hg = *(const v8bf*)(rp + 16);  // k = kbase+16+hi*8 .. +7
    return __builtin_shufflevector(lo, hg, 0, 1, 2, 3, 4, 5, 6, 7,
                                   8, 9, 10, 11, 12, 13, 14, 15);
  }
  v16bf f;
#pragma unroll
  for (int v = 0; v < 8; ++v) {
    int k = kbase + ((v >> 2) << 4) + (hi << 3) + ((v & 3) << 1);
    int k0 = k, k1 = k + 1;
    if (k0 > klimit - 1) k0 = klimit - 1;
    if (k1 > klimit - 1) k1 = klimit - 1;
    f[2 * v]     = p[(long)k0 * ld + r];
    f[2 * v + 1] = p[(long)k1 * ld + r];
  }
  return f;
}

// ---------------------------------------------------------------------------
// Generic GEMM:  D[M x N] = epi( A[M x K] (.) W[N x K]^T + bias, aux )
// One wave per 32x64 output tile (2x4 WMMA frags), K stepped by 32.
// A/W are bf16; bias/aux fp32; D (fp32) and/or Dbf (bf16) outputs.
// ---------------------------------------------------------------------------
template <bool TRA, bool TRB>
__global__ __launch_bounds__(256) void gemm_kernel(
    const bf16_t* __restrict__ A, long lda,
    const bf16_t* __restrict__ W, long ldw,
    const float* __restrict__ bias,
    const float* __restrict__ aux, long ldaux,
    float* __restrict__ D, bf16_t* __restrict__ Dbf, long ldd,
    int Mdim, int Ndim, int Kdim, int epi, float scale) {
  int wid = (int)((blockIdx.x * blockDim.x + threadIdx.x) >> 5);
  int ntm = (Mdim + 31) >> 5;
  int ntn = (Ndim + 63) >> 6;
  if (wid >= ntm * ntn) return;
  int tm = wid % ntm, tn = wid / ntm;
  int mbase = tm << 5, nbase = tn << 6;

  v8f acc[2][4];
#pragma unroll
  for (int i = 0; i < 2; ++i)
#pragma unroll
    for (int j = 0; j < 4; ++j)
#pragma unroll
      for (int v = 0; v < 8; ++v) acc[i][j][v] = 0.0f;

#pragma unroll 2
  for (int k0 = 0; k0 < Kdim; k0 += 32) {
    if (k0 + 32 < Kdim) {  // global_prefetch_b8 for the next K-chunk
      __builtin_prefetch(TRA ? &A[(long)(k0 + 32) * lda + mbase]
                             : &A[(long)mbase * lda + (k0 + 32)], 0, 1);
      __builtin_prefetch(TRB ? &W[(long)(k0 + 32) * ldw + nbase]
                             : &W[(long)nbase * ldw + (k0 + 32)], 0, 1);
    }
    v16bf a0 = load_frag_bf(A, lda, mbase,      k0, Mdim, Kdim, TRA);
    v16bf a1 = load_frag_bf(A, lda, mbase + 16, k0, Mdim, Kdim, TRA);
#pragma unroll
    for (int j = 0; j < 4; ++j) {
      v16bf wf = load_frag_bf(W, ldw, nbase + 16 * j, k0, Ndim, Kdim, TRB);
      acc[0][j] = wmma_bf16(a0, wf, acc[0][j]);
      acc[1][j] = wmma_bf16(a1, wf, acc[1][j]);
    }
  }

  int lane = threadIdx.x & 31;
  int nl = lane & 15, hi = lane >> 4;
#pragma unroll
  for (int i = 0; i < 2; ++i)
#pragma unroll
    for (int j = 0; j < 4; ++j) {
      int col = nbase + j * 16 + nl;
      if (col >= Ndim) continue;
      float bv = bias ? bias[col] : 0.0f;
#pragma unroll
      for (int v = 0; v < 8; ++v) {
        int row = mbase + i * 16 + v + (hi << 3);
        if (row >= Mdim) continue;
        float val = acc[i][j][v] + bv;
        switch (epi) {
          case EPI_SILU:  val = siluf(val); break;
          case EPI_GELU:  val = geluf(val); break;
          case EPI_RES:   val += aux[(long)row * ldaux + col]; break;
          case EPI_DPRED: val = (val - aux[(long)row * ldaux + col]) * scale; break;
          case EPI_DSILU: val = val * dsiluf(aux[(long)row * ldaux + col]); break;
          default: break;
        }
        if (D)   D[(long)row * ldd + col] = val;
        if (Dbf) Dbf[(long)row * ldd + col] = (bf16_t)val;
      }
    }
}

// ---------------------------------------------------------------------------
// Flash attention over bf16 qkv.  One wave per (b, h, 16-query tile); keys
// streamed 32 at a time.  S = Q K^T (4 WMMA), online softmax via half-wave
// shuffles, P staged through LDS (C-frag -> A-frag), O += P V (4 WMMA).
// ---------------------------------------------------------------------------
__global__ __launch_bounds__(256) void attn_kernel(const bf16_t* __restrict__ qkv,
                                                   bf16_t* __restrict__ y) {
  __shared__ bf16_t plds[8][16 * 32];
  const int TQ = (TA_ + 15) / 16;  // 66
  int wslot = (int)(threadIdx.x >> 5);
  int wid = (int)((blockIdx.x * blockDim.x + threadIdx.x) >> 5);
  if (wid >= B_ * H_ * TQ) return;
  int qt = wid % TQ;
  int bh = wid / TQ;
  int h = bh % H_, b = bh / H_;
  int qbase = qt * 16;
  int lane = threadIdx.x & 31;
  int nl = lane & 15, hi = lane >> 4;

  const long row3c = 3L * C_;
  const bf16_t* Qp = qkv + (long)b * TA_ * row3c + h * DH_;
  const bf16_t* Kp = Qp + C_;
  const bf16_t* Vp = Qp + 2 * C_;

  v16bf aQ0 = load_frag_bf(Qp, row3c, qbase, 0,  TA_, DH_, false);
  v16bf aQ1 = load_frag_bf(Qp, row3c, qbase, 32, TA_, DH_, false);

  v8f o[4];
#pragma unroll
  for (int j = 0; j < 4; ++j)
#pragma unroll
    for (int v = 0; v < 8; ++v) o[j][v] = 0.0f;
  float rmax[8], rsum[8];
#pragma unroll
  for (int v = 0; v < 8; ++v) { rmax[v] = -3.0e38f; rsum[v] = 0.0f; }

  for (int kb = 0; kb < TA_; kb += 32) {
    if (qbase >= PREFIX_ && kb > qbase + 15) break;  // fully-masked chunk

    v8f s0, s1;
#pragma unroll
    for (int v = 0; v < 8; ++v) { s0[v] = 0.0f; s1[v] = 0.0f; }
    v16bf kf;
    kf = load_frag_bf(Kp, row3c, kb,      0,  TA_, DH_, false); s0 = wmma_bf16(aQ0, kf, s0);
    kf = load_frag_bf(Kp, row3c, kb,      32, TA_, DH_, false); s0 = wmma_bf16(aQ1, kf, s0);
    kf = load_frag_bf(Kp, row3c, kb + 16, 0,  TA_, DH_, false); s1 = wmma_bf16(aQ0, kf, s1);
    kf = load_frag_bf(Kp, row3c, kb + 16, 32, TA_, DH_, false); s1 = wmma_bf16(aQ1, kf, s1);

#pragma unroll
    for (int v = 0; v < 8; ++v) {
      int q   = qbase + v + (hi << 3);
      int k0i = kb + nl;
      int k1i = kb + 16 + nl;
      float v0 = s0[v] * 0.125f;
      float v1 = s1[v] * 0.125f;
      bool m0 = (k0i >= TA_) || (q >= PREFIX_ && k0i >= PREFIX_ && k0i > q);
      bool m1 = (k1i >= TA_) || (q >= PREFIX_ && k1i >= PREFIX_ && k1i > q);
      s0[v] = m0 ? -3.0e38f : v0;
      s1[v] = m1 ? -3.0e38f : v1;
    }

#pragma unroll
    for (int v = 0; v < 8; ++v) {
      float m = fmaxf(s0[v], s1[v]);
#pragma unroll
      for (int off = 8; off; off >>= 1) m = fmaxf(m, __shfl_xor(m, off, 32));
      float mn    = fmaxf(rmax[v], m);
      float alpha = __expf(rmax[v] - mn);
      float p0 = (s0[v] < -1.0e37f) ? 0.0f : __expf(s0[v] - mn);
      float p1 = (s1[v] < -1.0e37f) ? 0.0f : __expf(s1[v] - mn);
      float ps = p0 + p1;
#pragma unroll
      for (int off = 8; off; off >>= 1) ps += __shfl_xor(ps, off, 32);
      rsum[v] = rsum[v] * alpha + ps;
      rmax[v] = mn;
      o[0][v] *= alpha; o[1][v] *= alpha; o[2][v] *= alpha; o[3][v] *= alpha;
      int prow = v + (hi << 3);
      plds[wslot][prow * 32 + nl]      = (bf16_t)p0;
      plds[wslot][prow * 32 + 16 + nl] = (bf16_t)p1;
    }
    // order the wave-local LDS store -> load (same wave, lockstep wave32)
    asm volatile("s_wait_dscnt 0" ::: "memory");

    v16bf aP;
#pragma unroll
    for (int v = 0; v < 8; ++v) {
      int k = ((v >> 2) << 4) + (hi << 3) + ((v & 3) << 1);
      aP[2 * v]     = plds[wslot][nl * 32 + k];
      aP[2 * v + 1] = plds[wslot][nl * 32 + k + 1];
    }
#pragma unroll
    for (int j = 0; j < 4; ++j) {
      v16bf vf = load_frag_bf(Vp, row3c, j * 16, kb, DH_, TA_, true);
      o[j] = wmma_bf16(aP, vf, o[j]);
    }
  }

  bf16_t* yb = y + (long)b * TA_ * C_ + h * DH_;
#pragma unroll
  for (int v = 0; v < 8; ++v) {
    int q = qbase + v + (hi << 3);
    if (q >= TA_) continue;
    float inv = 1.0f / rsum[v];
#pragma unroll
    for (int j = 0; j < 4; ++j)
      yb[(long)q * C_ + j * 16 + nl] = (bf16_t)(o[j][v] * inv);
  }
}

// ---------------------------------------------------------------------------
// Small support kernels
// ---------------------------------------------------------------------------
__global__ __launch_bounds__(256) void cvt_bf16_kernel(const float* __restrict__ in,
                                                       bf16_t* __restrict__ out, long n) {
  long i = (long)blockIdx.x * blockDim.x + threadIdx.x;
  if (i >= n) return;
  out[i] = (bf16_t)in[i];
}

// LayerNorm; fp32 in, bf16 out (only GEMMs consume LN outputs).
__global__ __launch_bounds__(256) void ln_kernel(const float* __restrict__ x,
                                                 const float* __restrict__ g,
                                                 const float* __restrict__ bta,
                                                 bf16_t* __restrict__ o, int C) {
  __shared__ float r1[256], r2[256];
  long row = blockIdx.x;
  const float* xr = x + row * (long)C;
  float a = 0.0f, q = 0.0f;
  for (int c = threadIdx.x; c < C; c += 256) { float v = xr[c]; a += v; q += v * v; }
  r1[threadIdx.x] = a; r2[threadIdx.x] = q;
  __syncthreads();
  for (int st = 128; st; st >>= 1) {
    if ((int)threadIdx.x < st) { r1[threadIdx.x] += r1[threadIdx.x + st]; r2[threadIdx.x] += r2[threadIdx.x + st]; }
    __syncthreads();
  }
  float mean = r1[0] / C;
  float var  = r2[0] / C - mean * mean;
  float inv  = rsqrtf(var + 1e-5f);
  for (int c = threadIdx.x; c < C; c += 256)
    o[row * (long)C + c] = (bf16_t)((xr[c] - mean) * inv * g[c] + bta[c]);
}

__global__ __launch_bounds__(256) void pool_kernel(const float* __restrict__ mem,
                                                   bf16_t* __restrict__ pooled) {
  long idx = (long)blockIdx.x * blockDim.x + threadIdx.x;
  long total = (long)B_ * MM_ * C_;
  if (idx >= total) return;
  int c = (int)(idx % C_);
  long r = idx / C_;
  int m = (int)(r % MM_);
  int b = (int)(r / MM_);
  const int G = T_ / MM_;  // 64
  const float* p = mem + ((long)b * T_ + (long)m * G) * C_ + c;
  float s = 0.0f;
  for (int j = 0; j < G; ++j) s += p[(long)j * C_];
  pooled[idx] = (bf16_t)(s * (1.0f / (float)G));
}

__global__ __launch_bounds__(256) void build_xa_kernel(const float* __restrict__ x,
                                                       const float* __restrict__ pm,
                                                       float* __restrict__ xa) {
  long idx = (long)blockIdx.x * blockDim.x + threadIdx.x;
  long total = (long)B_ * (PP_ + T_) * C_;
  if (idx >= total) return;
  int c = (int)(idx % C_);
  long r = idx / C_;
  int rr = (int)(r % (PP_ + T_));
  int b  = (int)(r / (PP_ + T_));
  float v = (rr < PP_) ? pm[(long)rr * C_ + c]
                       : x[((long)b * T_ + (rr - PP_)) * C_ + c];
  xa[((long)b * TA_ + MM_ + rr) * C_ + c] = v;
}

// out (fp32, final output) + bf16 copy for the keys/vals GEMMs.
__global__ __launch_bounds__(256) void copy_out_kernel(const float* __restrict__ xa3,
                                                       float* __restrict__ out,
                                                       bf16_t* __restrict__ outbf) {
  long idx = (long)blockIdx.x * blockDim.x + threadIdx.x;
  long total = (long)B_ * T_ * C_;
  if (idx >= total) return;
  int c = (int)(idx % C_);
  long r = idx / C_;
  int t = (int)(r % T_);
  int b = (int)(r / T_);
  float v = xa3[((long)b * TA_ + PREFIX_ + t) * C_ + c];
  out[idx] = v;
  outbf[idx] = (bf16_t)v;
}

__global__ __launch_bounds__(256) void silu_kernel(const float* __restrict__ in,
                                                   bf16_t* __restrict__ out, long n) {
  long i = (long)blockIdx.x * blockDim.x + threadIdx.x;
  if (i >= n) return;
  out[i] = (bf16_t)siluf(in[i]);
}

__global__ __launch_bounds__(256) void colsum_kernel(const float* __restrict__ src,
                                                     float* __restrict__ dst,
                                                     int rows, int cols, int nb) {
  long idx = (long)blockIdx.x * blockDim.x + threadIdx.x;
  long total = (long)nb * cols;
  if (idx >= total) return;
  int b = (int)(idx / cols);
  int c = (int)(idx % cols);
  const float* p = src + (long)b * rows * cols + c;
  float s = 0.0f;
  for (int r = 0; r < rows; ++r) s += p[(long)r * cols];
  dst[idx] = s;
}

__global__ __launch_bounds__(256) void sgd_kernel(const float* __restrict__ w,
                                                  const float* __restrict__ mom,
                                                  const float* __restrict__ g,
                                                  float* __restrict__ nw,
                                                  float* __restrict__ nmom, long n) {
  long i = (long)blockIdx.x * blockDim.x + threadIdx.x;
  if (i >= n) return;
  float nm = MU_ * mom[i] + g[i];
  nmom[i] = nm;
  nw[i] = (1.0f - DECAY_) * w[i] - LR_ * nm;
}

// ---------------------------------------------------------------------------
// Host-side launch helpers
// ---------------------------------------------------------------------------
static inline void launch_gemm(hipStream_t st, bool tra, bool trb, int epi,
                               const bf16_t* A, long lda, const bf16_t* W, long ldw,
                               const float* bias, const float* aux, long ldaux,
                               float* D, bf16_t* Dbf, long ldd, int M, int N, int K,
                               float scale = 1.0f) {
  long waves = (long)((M + 31) / 32) * ((N + 63) / 64);
  int blocks = (int)((waves + 7) / 8);
  dim3 g(blocks), blk(256);
  if (!tra && !trb)
    gemm_kernel<false, false><<<g, blk, 0, st>>>(A, lda, W, ldw, bias, aux, ldaux, D, Dbf, ldd, M, N, K, epi, scale);
  else if (!tra && trb)
    gemm_kernel<false, true><<<g, blk, 0, st>>>(A, lda, W, ldw, bias, aux, ldaux, D, Dbf, ldd, M, N, K, epi, scale);
  else if (tra && !trb)
    gemm_kernel<true, false><<<g, blk, 0, st>>>(A, lda, W, ldw, bias, aux, ldaux, D, Dbf, ldd, M, N, K, epi, scale);
  else
    gemm_kernel<true, true><<<g, blk, 0, st>>>(A, lda, W, ldw, bias, aux, ldaux, D, Dbf, ldd, M, N, K, epi, scale);
}

static inline int ew_blocks(long n) { return (int)((n + 255) / 256); }

extern "C" void kernel_launch(void* const* d_in, const int* in_sizes, int n_in,
                              void* d_out, int out_size, void* d_ws, size_t ws_size,
                              hipStream_t stream) {
  (void)in_sizes; (void)n_in; (void)out_size; (void)ws_size;
  const float* x           = (const float*)d_in[0];
  const float* persist_mem = (const float*)d_in[1];
  const float* ln1_w       = (const float*)d_in[2];
  const float* ln1_b       = (const float*)d_in[3];
  const float* ln2_w       = (const float*)d_in[4];
  const float* ln2_b       = (const float*)d_in[5];
  const float* attn_w      = (const float*)d_in[6];
  const float* attn_b      = (const float*)d_in[7];
  const float* attn_proj_w = (const float*)d_in[8];
  const float* attn_proj_b = (const float*)d_in[9];
  const float* fc_w        = (const float*)d_in[10];
  const float* fc_b        = (const float*)d_in[11];
  const float* mlp_proj_w  = (const float*)d_in[12];
  const float* mlp_proj_b  = (const float*)d_in[13];
  const float* q_w         = (const float*)d_in[14];
  const float* q_b         = (const float*)d_in[15];
  const float* k_w         = (const float*)d_in[16];
  const float* k_b         = (const float*)d_in[17];
  const float* v_w         = (const float*)d_in[18];
  const float* v_b         = (const float*)d_in[19];
  const float* o_w         = (const float*)d_in[20];
  const float* o_b         = (const float*)d_in[21];
  const float* mem_w0      = (const float*)d_in[22];
  const float* mem_b0      = (const float*)d_in[23];
  const float* mem_w1      = (const float*)d_in[24];
  const float* mem_b1      = (const float*)d_in[25];
  const float* mom_w0      = (const float*)d_in[26];
  const float* mom_b0      = (const float*)d_in[27];
  const float* mom_w1      = (const float*)d_in[28];
  const float* mom_b1      = (const float*)d_in[29];

  float* out = (float*)d_out;

  // ---- output layout (tuple return order) ----
  const long BTC = (long)B_ * T_ * C_;
  const long BHC = (long)B_ * HID_ * C_;
  long oo = 0;
  float* o_out = out + oo; oo += BTC;
  float* o_nw0 = out + oo; oo += BHC;
  float* o_nb0 = out + oo; oo += (long)B_ * HID_;
  float* o_nw1 = out + oo; oo += (long)B_ * C_ * HID_;
  float* o_nb1 = out + oo; oo += (long)B_ * C_;
  float* o_mw0 = out + oo; oo += BHC;
  float* o_mb0 = out + oo; oo += (long)B_ * HID_;
  float* o_mw1 = out + oo; oo += (long)B_ * C_ * HID_;
  float* o_mb1 = out + oo;

  // ---- workspace bump allocator ----
  char* wsb = (char*)d_ws;
  size_t off = 0;
  auto allocf = [&](long n) -> float* {
    float* p = (float*)(wsb + off);
    off += ((size_t)n * 4 + 255) & ~(size_t)255;
    return p;
  };
  auto allocb = [&](long n) -> bf16_t* {
    bf16_t* p = (bf16_t*)(wsb + off);
    off += ((size_t)n * 2 + 255) & ~(size_t)255;
    return p;
  };
  // fp32 intermediates (residual streams / epilogue aux / reductions)
  float* membuf = allocf(BTC);
  float* xa     = allocf((long)B_ * TA_ * C_);
  float* xa2    = allocf((long)B_ * TA_ * C_);
  float* xa3    = allocf((long)B_ * TA_ * C_);
  float* vals   = allocf(BTC);
  float* hhpre  = allocf((long)B_ * T_ * HID_);
  float* dpredf = allocf(BTC);
  float* dhpref = allocf((long)B_ * T_ * HID_);
  float* gw0    = allocf(BHC);
  float* gw1    = allocf((long)B_ * C_ * HID_);
  float* gb0    = allocf((long)B_ * HID_);
  float* gb1    = allocf((long)B_ * C_);
  // bf16 GEMM operands
  bf16_t* xbf     = allocb(BTC);
  bf16_t* wq_bf   = allocb((long)C_ * C_);
  bf16_t* wk_bf   = allocb((long)C_ * C_);
  bf16_t* wv_bf   = allocb((long)C_ * C_);
  bf16_t* wo_bf   = allocb((long)C_ * C_);
  bf16_t* wpj_bf  = allocb((long)C_ * C_);
  bf16_t* watt_bf = allocb(3L * C_ * C_);
  bf16_t* wfc_bf  = allocb(4L * C_ * C_);
  bf16_t* wml_bf  = allocb(4L * C_ * C_);
  bf16_t* w0bf    = allocb(BHC);
  bf16_t* w1bf    = allocb((long)B_ * C_ * HID_);
  bf16_t* qbf     = allocb(BTC);
  bf16_t* hbf     = allocb((long)B_ * T_ * HID_);
  bf16_t* pooledb = allocb((long)B_ * MM_ * C_);
  bf16_t* hlnbf   = allocb((long)B_ * TA_ * C_);
  bf16_t* qkvbf   = allocb((long)B_ * TA_ * 3 * C_);
  bf16_t* ybf     = allocb((long)B_ * TA_ * C_);
  bf16_t* h2bf    = allocb((long)B_ * TA_ * C_);
  bf16_t* fcbf    = allocb((long)B_ * TA_ * 4 * C_);
  bf16_t* outbf   = allocb(BTC);
  bf16_t* keysbf  = allocb(BTC);
  bf16_t* hhbf    = allocb((long)B_ * T_ * HID_);
  bf16_t* dpredb  = allocb(BTC);
  bf16_t* dhpreb  = allocb((long)B_ * T_ * HID_);

  // ---- stage bf16 operands (one pass; GEMMs then read pure bf16) ----
  auto cvt = [&](const float* src, bf16_t* dst, long n) {
    cvt_bf16_kernel<<<ew_blocks(n), 256, 0, stream>>>(src, dst, n);
  };
  cvt(x, xbf, BTC);
  cvt(q_w, wq_bf, (long)C_ * C_);
  cvt(k_w, wk_bf, (long)C_ * C_);
  cvt(v_w, wv_bf, (long)C_ * C_);
  cvt(o_w, wo_bf, (long)C_ * C_);
  cvt(attn_proj_w, wpj_bf, (long)C_ * C_);
  cvt(attn_w, watt_bf, 3L * C_ * C_);
  cvt(fc_w, wfc_bf, 4L * C_ * C_);
  cvt(mlp_proj_w, wml_bf, 4L * C_ * C_);
  cvt(mem_w0, w0bf, BHC);
  cvt(mem_w1, w1bf, (long)B_ * C_ * HID_);

  // ================== forward: neural-memory read ==================
  launch_gemm(stream, false, false, EPI_NONE, xbf, C_, wq_bf, C_, q_b, nullptr, 0,
              nullptr, qbf, C_, B_ * T_, C_, C_);
  for (int b = 0; b < B_; ++b)
    launch_gemm(stream, false, false, EPI_SILU, qbf + (long)b * T_ * C_, C_,
                w0bf + (long)b * HID_ * C_, C_, mem_b0 + (long)b * HID_,
                nullptr, 0, nullptr, hbf + (long)b * T_ * HID_, HID_, T_, HID_, C_);
  for (int b = 0; b < B_; ++b)
    launch_gemm(stream, false, false, EPI_NONE, hbf + (long)b * T_ * HID_, HID_,
                w1bf + (long)b * C_ * HID_, HID_, mem_b1 + (long)b * C_,
                nullptr, 0, membuf + (long)b * T_ * C_, nullptr, C_, T_, C_, HID_);
  pool_kernel<<<ew_blocks((long)B_ * MM_ * C_), 256, 0, stream>>>(membuf, pooledb);
  for (int b = 0; b < B_; ++b)
    launch_gemm(stream, false, false, EPI_NONE, pooledb + (long)b * MM_ * C_, C_,
                wo_bf, C_, o_b, nullptr, 0, xa + (long)b * TA_ * C_, nullptr, C_,
                MM_, C_, C_);
  build_xa_kernel<<<ew_blocks((long)B_ * (PP_ + T_) * C_), 256, 0, stream>>>(x, persist_mem, xa);

  // ================== transformer block ==================
  ln_kernel<<<B_ * TA_, 256, 0, stream>>>(xa, ln1_w, ln1_b, hlnbf, C_);
  launch_gemm(stream, false, false, EPI_NONE, hlnbf, C_, watt_bf, C_, attn_b,
              nullptr, 0, nullptr, qkvbf, 3 * C_, B_ * TA_, 3 * C_, C_);
  {
    const int TQ = (TA_ + 15) / 16;
    long waves = (long)B_ * H_ * TQ;
    attn_kernel<<<(int)((waves + 7) / 8), 256, 0, stream>>>(qkvbf, ybf);
  }
  launch_gemm(stream, false, false, EPI_RES, ybf, C_, wpj_bf, C_, attn_proj_b,
              xa, C_, xa2, nullptr, C_, B_ * TA_, C_, C_);
  ln_kernel<<<B_ * TA_, 256, 0, stream>>>(xa2, ln2_w, ln2_b, h2bf, C_);
  launch_gemm(stream, false, false, EPI_GELU, h2bf, C_, wfc_bf, C_, fc_b,
              nullptr, 0, nullptr, fcbf, 4 * C_, B_ * TA_, 4 * C_, C_);
  launch_gemm(stream, false, false, EPI_RES, fcbf, 4 * C_, wml_bf, 4 * C_,
              mlp_proj_b, xa2, C_, xa3, nullptr, C_, B_ * TA_, C_, 4 * C_);
  copy_out_kernel<<<ew_blocks(BTC), 256, 0, stream>>>(xa3, o_out, outbf);

  // ================== memory update (TTT step) ==================
  launch_gemm(stream, false, false, EPI_NONE, outbf, C_, wk_bf, C_, k_b,
              nullptr, 0, nullptr, keysbf, C_, B_ * T_, C_, C_);
  launch_gemm(stream, false, false, EPI_NONE, outbf, C_, wv_bf, C_, v_b,
              nullptr, 0, vals, nullptr, C_, B_ * T_, C_, C_);
  for (int b = 0; b < B_; ++b) {
    const bf16_t* w0b = w0bf + (long)b * HID_ * C_;
    const bf16_t* w1b = w1bf + (long)b * C_ * HID_;
    const bf16_t* kb_ = keysbf + (long)b * T_ * C_;
    const float*  vb_ = vals + (long)b * T_ * C_;
    float*  hpb = hhpre + (long)b * T_ * HID_;
    bf16_t* hhb = hhbf  + (long)b * T_ * HID_;
    // hh_pre = keys @ w0^T + b0
    launch_gemm(stream, false, false, EPI_NONE, kb_, C_, w0b, C_,
                mem_b0 + (long)b * HID_, nullptr, 0, hpb, nullptr, HID_, T_, HID_, C_);
    silu_kernel<<<ew_blocks((long)T_ * HID_), 256, 0, stream>>>(hpb, hhb, (long)T_ * HID_);
    // dpred = (hh @ w1^T + b1 - vals) * 2/(T*C)
    launch_gemm(stream, false, false, EPI_DPRED, hhb, HID_, w1b, HID_,
                mem_b1 + (long)b * C_, vb_, C_, dpredf + (long)b * T_ * C_,
                dpredb + (long)b * T_ * C_, C_, T_, C_, HID_,
                2.0f / (float)(T_ * C_));
    // g_w1 = dpred^T @ hh        (C x HID)
    launch_gemm(stream, true, true, EPI_NONE, dpredb + (long)b * T_ * C_, C_,
                hhb, HID_, nullptr, nullptr, 0,
                gw1 + (long)b * C_ * HID_, nullptr, HID_, C_, HID_, T_);
    // dhpre = (dpred @ w1) * silu'(hh_pre)
    launch_gemm(stream, false, true, EPI_DSILU, dpredb + (long)b * T_ * C_, C_,
                w1b, HID_, nullptr, hpb, HID_, dhpref + (long)b * T_ * HID_,
                dhpreb + (long)b * T_ * HID_, HID_, T_, HID_, C_);
    // g_w0 = dhpre^T @ keys      (HID x C)
    launch_gemm(stream, true, true, EPI_NONE, dhpreb + (long)b * T_ * HID_, HID_,
                kb_, C_, nullptr, nullptr, 0,
                gw0 + (long)b * HID_ * C_, nullptr, C_, HID_, C_, T_);
  }
  colsum_kernel<<<ew_blocks((long)B_ * C_), 256, 0, stream>>>(dpredf, gb1, T_, C_, B_);
  colsum_kernel<<<ew_blocks((long)B_ * HID_), 256, 0, stream>>>(dhpref, gb0, T_, HID_, B_);

  // momentum + weight-decay updates -> outputs
  sgd_kernel<<<ew_blocks(BHC), 256, 0, stream>>>(mem_w0, mom_w0, gw0, o_nw0, o_mw0, BHC);
  sgd_kernel<<<ew_blocks((long)B_ * HID_), 256, 0, stream>>>(mem_b0, mom_b0, gb0, o_nb0, o_mb0, (long)B_ * HID_);
  sgd_kernel<<<ew_blocks((long)B_ * C_ * HID_), 256, 0, stream>>>(mem_w1, mom_w1, gw1, o_nw1, o_mw1, (long)B_ * C_ * HID_);
  sgd_kernel<<<ew_blocks((long)B_ * C_), 256, 0, stream>>>(mem_b1, mom_b1, gb1, o_nb1, o_mb1, (long)B_ * C_);
}
